// OGNIDC_40338332844087
// MI455X (gfx1250) — compile-verified, matrix operand order checked by reading, not run on hardware
//
#include <hip/hip_runtime.h>
#include <hip/hip_bf16.h>
#include <stdint.h>

// ---------------- CDNA5 types ----------------
typedef __bf16 bf16_t;
typedef __attribute__((ext_vector_type(16))) __bf16 v16bf;
typedef __attribute__((ext_vector_type(8)))  float  v8f;
typedef __attribute__((ext_vector_type(4)))  unsigned int v4u;
typedef __attribute__((ext_vector_type(8)))  int v8i;
typedef __attribute__((ext_vector_type(4)))  int v4i;

#define B_SZ 4
#define H_FULL 352
#define W_FULL 1216
#define H_DS 44
#define W_DS 152
#define HW_DS (H_DS * W_DS)      // 6688
#define GRU_ITERS 3
#define ALPHA_C 1.0f
#define CLAMP_C 10.0f
#define CONF_MIN_C 0.01f
#define CG_ITERS_C 64

#if defined(__has_builtin)
#  if __has_builtin(__builtin_amdgcn_tensor_load_to_lds)
#    define HAVE_TDM 1
#  endif
#  if __has_builtin(__builtin_amdgcn_s_wait_tensorcnt)
#    define HAVE_TCNT 1
#  endif
#endif
#ifndef HAVE_TDM
#  define HAVE_TDM 0
#endif
#ifndef HAVE_TCNT
#  define HAVE_TCNT 0
#endif

// ---------------- helpers ----------------
__device__ inline bf16_t f2bf(float f) {
  unsigned u = __builtin_bit_cast(unsigned, f);
  unsigned r = (u + 0x7FFFu + ((u >> 16) & 1u)) >> 16;
  unsigned short s = (unsigned short)r;
  return __builtin_bit_cast(bf16_t, s);
}
__device__ inline float sigmoidf_(float v) { return 1.f / (1.f + __expf(-v)); }

template <int ACT> __device__ inline float act_f(float v) {
  if (ACT == 1) return fmaxf(v, 0.f);
  if (ACT == 2) return sigmoidf_(v);
  if (ACT == 3) return tanhf(v);
  return v;
}

__device__ inline float block_reduce_sum(float v, float* red) {
  __syncthreads();
#pragma unroll
  for (int off = 16; off > 0; off >>= 1) v += __shfl_down(v, off, 32);
  const int lane = threadIdx.x & 31, wid = threadIdx.x >> 5;
  if (lane == 0) red[wid] = v;
  __syncthreads();
  const int nw = (blockDim.x + 31) >> 5;
  if (threadIdx.x < 32) {
    float t = (threadIdx.x < (unsigned)nw) ? red[threadIdx.x] : 0.f;
#pragma unroll
    for (int off = 16; off > 0; off >>= 1) t += __shfl_down(t, off, 32);
    if (threadIdx.x == 0) red[0] = t;
  }
  __syncthreads();
  return red[0];
}

// Fragment K-permutation within a 32-wide K block (ISA §7.12.2 16-bit layouts):
// lane half hl, element e  <->  k32 = (e<8) ? e+8*hl : 8+e+8*hl.
// Stored position within block: pos = hl*16 + e  (so each fragment is 16 contiguous bf16).
__device__ __host__ inline int kperm_fwd(int pos) {   // pos -> k32
  int hl = pos >> 4, e = pos & 15;
  return (e < 8) ? (e + 8 * hl) : (8 + e + 8 * hl);
}

// ---------------- TDM weight-tile staging (ISA cdna5 §8: D# groups) ----------------
#if HAVE_TDM && defined(__gfx1250__)
__device__ inline void tdm_load_tile_bf16(unsigned lds_byte_addr, const bf16_t* gptr,
                                          unsigned tensor_d0, unsigned tensor_d1,
                                          unsigned tile_d0, unsigned tile_d1,
                                          unsigned stride_elems) {
  // Approximate D# packing per ISA §8.3/8.4: group0 = {count, lds_addr, global_addr, type=2},
  // group1 = {data_size=2B, tensor_dim0/1, tile_dim0/1, tensor_dim0_stride}; 2D tensor,
  // groups 2/3 zero (NULL-equivalent payload).
  unsigned long long ga = (unsigned long long)(uintptr_t)gptr;
  v4u g0;
  g0[0] = 1u;                                                   // count=1 user D#
  g0[1] = lds_byte_addr;
  g0[2] = (unsigned)ga;
  g0[3] = (unsigned)((ga >> 32) & 0x01FFFFFFull) | (2u << 30);  // addr[56:32] | type=2
  v8i g1;
  g1[0] = (1 << 16);                                            // data_size=1 (2 bytes)
  g1[1] = (int)((tensor_d0 & 0xFFFFu) << 16);
  g1[2] = (int)((tensor_d0 >> 16) & 0xFFFFu) | (int)((tensor_d1 & 0xFFFFu) << 16);
  g1[3] = (int)((tensor_d1 >> 16) & 0xFFFFu) | (int)((tile_d0 & 0xFFFFu) << 16);
  g1[4] = (int)(tile_d1 & 0xFFFFu);
  g1[5] = (int)(stride_elems << 16);
  g1[6] = (int)(stride_elems >> 16);
  g1[7] = 0;
  v4i gz = {0, 0, 0, 0};
#if __clang_major__ >= 23
  v8i gz8 = {0, 0, 0, 0, 0, 0, 0, 0};
  __builtin_amdgcn_tensor_load_to_lds(g0, g1, gz, gz, gz8, 0);
#else
  __builtin_amdgcn_tensor_load_to_lds(g0, g1, gz, gz, 0);
#endif
}
#endif

// ---------------- implicit-GEMM conv 3x3 SAME, bf16 WMMA, f32 accum ----------------
// GEMM view (per batch): M = Hout*Wout pixels, N = Cout, K = Cin*9 (padded to 64).
// Block = 256 threads (8 waves); block tile M=128 x N=64; K staged 64 per round.
// Wave wm (0..7) owns rows [wm*16, wm*16+16) x all 64 N -> 4 v8f accumulators.
// LDS is fragment-major: every lane fragment (A or B) is 16 contiguous bf16 (32 B)
// -> ds_load_b128-class vector loads instead of scalar gathers.
template <int ACT>
__global__ __launch_bounds__(256) void k_conv_wmma(
    const float* __restrict__ in, const bf16_t* __restrict__ Wp,
    const float* __restrict__ bias, float* __restrict__ out,
    int Cin, int Cout, int Hin, int Win, int Hout, int Wout,
    int stride, int padlo, int Kpad) {
  const int K = Cin * 9;
  const int M = Hout * Wout;
  const int m0 = blockIdx.x * 128;
  const int n0 = blockIdx.y * 64;
  const int bidx = blockIdx.z;

  __shared__ __attribute__((aligned(32))) bf16_t As[128 * 64];  // [m][k'] frag-major, 16 KB
  __shared__ __attribute__((aligned(32))) bf16_t Bs[64 * 64];   // [n][k'] frag-major, 8 KB

  const float* inB = in + (size_t)bidx * Cin * Hin * Win;
  const int tid = threadIdx.x;
  const int lane = tid & 31;
  const int wm = tid >> 5;        // wave id = M-row group
  const int hl = lane >> 4;       // lane half (fragment K-half select)
  const int row = lane & 15;

  v8f acc0 = {}, acc1 = {}, acc2 = {}, acc3 = {};

  for (int kt = 0; kt < Kpad; kt += 64) {
    // ---- stage A: 128x64 im2col tile, written in fragment-major permuted order ----
    for (int idx = tid; idx < 128 * 64; idx += 256) {
      const int mm = idx >> 6, t = idx & 63;
      const int sub = t >> 5, pos = t & 31;
      const int k = kt + sub * 32 + kperm_fwd(pos);
      const int m = m0 + mm;
      float v = 0.f;
      if (m < M && k < K) {
        const int ci = k / 9, rem = k - ci * 9;
        const int ky = rem / 3, kx = rem - ky * 3;
        const int oy = m / Wout, ox = m - oy * Wout;
        const int iy = oy * stride + ky - padlo;
        const int ix = ox * stride + kx - padlo;
        if ((unsigned)iy < (unsigned)Hin && (unsigned)ix < (unsigned)Win)
          v = inB[((size_t)ci * Hin + iy) * Win + ix];
      }
      As[idx] = f2bf(v);
    }
    // ---- stage B: 64 rows (n) x 64 k' (already fragment-permuted + zero-padded in
    //      global memory), plain row-major 2D tile -> Tensor Data Mover ----
#if HAVE_TDM && defined(__gfx1250__)
    if (wm == 0) {
      const unsigned lds_addr = (unsigned)(uintptr_t)(&Bs[0]);
      tdm_load_tile_bf16(lds_addr, Wp + (size_t)n0 * Kpad + kt,
                         (unsigned)Kpad, (unsigned)Cout, 64u, 64u, (unsigned)Kpad);
#if HAVE_TCNT
      __builtin_amdgcn_s_wait_tensorcnt(0);
#endif
    }
#else
    for (int idx = tid; idx < 64 * 64; idx += 256) {
      const int nn = idx >> 6, kk = idx & 63;
      const int n = n0 + nn;
      Bs[idx] = (n < Cout) ? Wp[(size_t)n * Kpad + kt + kk] : f2bf(0.f);
    }
#endif
    if (lane == 0 && kt + 64 < Kpad)
      __builtin_prefetch(Wp + (size_t)n0 * Kpad + kt + 64, 0, 1);  // global_prefetch_b8
    __syncthreads();

    // ---- 2 K-substeps x 4 N-tiles: 8 WMMAs per staged round ----
#pragma unroll
    for (int sub = 0; sub < 2; ++sub) {
      const int fo = sub * 32 + hl * 16;  // fragment offset within a 64-wide k' row
      const v16bf af = *(const v16bf*)&As[(wm * 16 + row) * 64 + fo];
      const v16bf b0 = *(const v16bf*)&Bs[(row) * 64 + fo];
      const v16bf b1 = *(const v16bf*)&Bs[(16 + row) * 64 + fo];
      const v16bf b2 = *(const v16bf*)&Bs[(32 + row) * 64 + fo];
      const v16bf b3 = *(const v16bf*)&Bs[(48 + row) * 64 + fo];
      acc0 = __builtin_amdgcn_wmma_f32_16x16x32_bf16(false, af, false, b0, (short)0, acc0, false, false);
      acc1 = __builtin_amdgcn_wmma_f32_16x16x32_bf16(false, af, false, b1, (short)0, acc1, false, false);
      acc2 = __builtin_amdgcn_wmma_f32_16x16x32_bf16(false, af, false, b2, (short)0, acc2, false, false);
      acc3 = __builtin_amdgcn_wmma_f32_16x16x32_bf16(false, af, false, b3, (short)0, acc3, false, false);
    }
    __syncthreads();
  }

  // ---- epilogue: C/D 16x16 f32 layout: VGPR i -> M = i + 8*hl, N = lane&15 ----
#pragma unroll
  for (int i = 0; i < 8; ++i) {
    const int m = m0 + wm * 16 + hl * 8 + i;
    if (m >= M) continue;
    const size_t obase = (size_t)bidx * Cout;
#pragma unroll
    for (int j = 0; j < 4; ++j) {
      const int n = n0 + j * 16 + row;
      if (n < Cout) {
        const float a = (j == 0) ? acc0[i] : (j == 1) ? acc1[i] : (j == 2) ? acc2[i] : acc3[i];
        out[(obase + n) * M + m] = act_f<ACT>(a + bias[n]);
      }
    }
  }
}

// ---------------- weight convert: f32 OIHW -> bf16, fragment-permuted, K padded ----------------
__global__ void k_wperm(const float* __restrict__ in, bf16_t* __restrict__ out,
                        int Cout, int K, int Kpad) {
  int i = blockIdx.x * blockDim.x + threadIdx.x;
  if (i >= Cout * Kpad) return;
  const int n = i / Kpad, kp = i - n * Kpad;
  const int blk = kp >> 5, pos = kp & 31;
  const int k = blk * 32 + kperm_fwd(pos);
  out[i] = (k < K) ? f2bf(in[(size_t)n * K + k]) : f2bf(0.f);
}

// ---------------- small kernels ----------------
__global__ void k_zero(float* __restrict__ p, int n) {
  int i = blockIdx.x * blockDim.x + threadIdx.x;
  if (i < n) p[i] = 0.f;
}

__global__ void k_concat4(const float* __restrict__ rgb, const float* __restrict__ dep,
                          float* __restrict__ x0, int hw) {
  int i = blockIdx.x * blockDim.x + threadIdx.x;
  int total = B_SZ * 4 * hw;
  if (i >= total) return;
  int j = i % hw, c = (i / hw) & 3, b = i / (4 * hw);
  x0[i] = (c < 3) ? rgb[((size_t)b * 3 + c) * hw + j] : dep[(size_t)b * hw + j];
}

__global__ void k_downsample(const float* __restrict__ dep, float* __restrict__ mds,
                             float* __restrict__ slog) {
  int i = blockIdx.x * blockDim.x + threadIdx.x;
  if (i >= B_SZ * HW_DS) return;
  int b = i / HW_DS, r = i % HW_DS, y = r / W_DS, x = r % W_DS;
  const float* d = dep + (size_t)b * H_FULL * W_FULL;
  float sum = 0.f; int cnt = 0;
  for (int dy = 0; dy < 8; ++dy)
    for (int dx = 0; dx < 8; ++dx) {
      float v = d[(size_t)(y * 8 + dy) * W_FULL + (x * 8 + dx)];
      sum += v;
      cnt += (v > 0.f);
    }
  float mean_d = sum * (1.f / 64.f);
  float mfrac = (float)cnt * (1.f / 64.f);
  float ds = (mfrac > 0.f) ? (mean_d / fmaxf(mfrac, 1e-8f)) : 0.f;
  float mb = (mfrac > 0.f) ? 1.f : 0.f;
  mds[i] = mb;
  slog[i] = (mb > 0.f) ? logf(fmaxf(ds, 1e-8f)) : 0.f;
}

__global__ void k_splitctx(const float* __restrict__ ctx, float* __restrict__ net,
                           float* __restrict__ inp) {
  int i = blockIdx.x * blockDim.x + threadIdx.x;
  if (i >= B_SZ * 128 * HW_DS) return;
  int j = i % HW_DS, c = (i / HW_DS) % 128, b = i / (128 * HW_DS);
  net[i] = tanhf(ctx[((size_t)b * 256 + c) * HW_DS + j]);
  inp[i] = fmaxf(ctx[((size_t)b * 256 + 128 + c) * HW_DS + j], 0.f);
}

__global__ void k_wa(const float* __restrict__ mds, const float* __restrict__ conf,
                     float* __restrict__ wa, int n) {
  int i = blockIdx.x * blockDim.x + threadIdx.x;
  if (i < n) wa[i] = ALPHA_C * mds[i] * conf[i];
}

__global__ void k_mean(const float* __restrict__ ld, float* __restrict__ meanb) {
  __shared__ float red[32];
  int b = blockIdx.x;
  float s = 0.f;
  for (int i = threadIdx.x; i < HW_DS; i += blockDim.x) s += ld[(size_t)b * HW_DS + i];
  s = block_reduce_sum(s, red);
  if (threadIdx.x == 0) meanb[b] = s / (float)HW_DS;
}

__global__ void k_mencin(const float* __restrict__ ld, const float* __restrict__ g,
                         const float* __restrict__ meanb, float* __restrict__ out) {
  int i = blockIdx.x * blockDim.x + threadIdx.x;
  if (i >= B_SZ * HW_DS) return;
  int b = i / HW_DS, j = i % HW_DS;
  out[((size_t)b * 3 + 0) * HW_DS + j] = ld[i] - meanb[b];
  out[((size_t)b * 3 + 1) * HW_DS + j] = g[((size_t)b * 2 + 0) * HW_DS + j];
  out[((size_t)b * 3 + 2) * HW_DS + j] = g[((size_t)b * 2 + 1) * HW_DS + j];
}

__global__ void k_hx(const float* __restrict__ net, const float* __restrict__ inp,
                     const float* __restrict__ mfeat, float* __restrict__ hx) {
  int i = blockIdx.x * blockDim.x + threadIdx.x;
  if (i >= B_SZ * 320 * HW_DS) return;
  int j = i % HW_DS, c = (i / HW_DS) % 320, b = i / (320 * HW_DS);
  float v;
  if (c < 128)       v = net[((size_t)b * 128 + c) * HW_DS + j];
  else if (c < 256)  v = inp[((size_t)b * 128 + (c - 128)) * HW_DS + j];
  else               v = mfeat[((size_t)b * 64 + (c - 256)) * HW_DS + j];
  hx[i] = v;
}

__global__ void k_rnet(float* __restrict__ hx, const float* __restrict__ r,
                       const float* __restrict__ net) {
  int i = blockIdx.x * blockDim.x + threadIdx.x;
  if (i >= B_SZ * 128 * HW_DS) return;
  int j = i % HW_DS, c = (i / HW_DS) % 128, b = i / (128 * HW_DS);
  hx[((size_t)b * 320 + c) * HW_DS + j] = r[i] * net[i];
}

__global__ void k_gru(float* __restrict__ net, const float* __restrict__ z,
                      const float* __restrict__ q, int n) {
  int i = blockIdx.x * blockDim.x + threadIdx.x;
  if (i < n) net[i] = (1.f - z[i]) * net[i] + z[i] * q[i];
}

__global__ void k_post(float* __restrict__ g, const float* __restrict__ dg,
                       float* __restrict__ cf, int n) {
  int i = blockIdx.x * blockDim.x + threadIdx.x;
  if (i < n) {
    g[i] = fminf(fmaxf(g[i] + dg[i], -CLAMP_C), CLAMP_C);
    cf[i] = CONF_MIN_C + (1.f - CONF_MIN_C) * sigmoidf_(cf[i]);
  }
}

// b = DxT(cf0*g0) + DyT(cf1*g1) + wa*sparse_log
__global__ void k_buildb(const float* __restrict__ cf, const float* __restrict__ g,
                         const float* __restrict__ wa, const float* __restrict__ slog,
                         float* __restrict__ bvec) {
  int i = blockIdx.x * blockDim.x + threadIdx.x;
  if (i >= B_SZ * HW_DS) return;
  int b = i / HW_DS, r = i % HW_DS, y = r / W_DS, x = r % W_DS;
  const float* cf0 = cf + (size_t)b * 2 * HW_DS;
  const float* cf1 = cf0 + HW_DS;
  const float* g0 = g + (size_t)b * 2 * HW_DS;
  const float* g1 = g0 + HW_DS;
  float acc = wa[i] * slog[i];
  if (x > 0)          acc += cf0[r - 1] * g0[r - 1];
  if (x < W_DS - 1)   acc -= cf0[r] * g0[r];
  if (y > 0)          acc += cf1[r - W_DS] * g1[r - W_DS];
  if (y < H_DS - 1)   acc -= cf1[r] * g1[r];
  bvec[i] = acc;
}

// ---------------- CG solve: 1 workgroup per batch, all 64 iters in LDS ----------------
__device__ inline float A_apply(const float* __restrict__ d, const float* __restrict__ cx,
                                const float* __restrict__ cy, const float* __restrict__ wa,
                                int idx) {
  const int y = idx / W_DS, x = idx - y * W_DS;
  const float dc = d[idx];
  float acc = wa[idx] * dc;
  if (x > 0)          acc += cx[idx - 1] * (dc - d[idx - 1]);
  if (x < W_DS - 1)   acc -= cx[idx] * (d[idx + 1] - dc);
  if (y > 0)          acc += cy[idx - W_DS] * (dc - d[idx - W_DS]);
  if (y < H_DS - 1)   acc -= cy[idx] * (d[idx + W_DS] - dc);
  return acc;
}

__global__ __launch_bounds__(1024) void k_cg(const float* __restrict__ cf,
                                             const float* __restrict__ wa,
                                             const float* __restrict__ bvec,
                                             float* __restrict__ ld, int iters) {
  constexpr int EP = (HW_DS + 1023) / 1024;  // 7
  __shared__ float pbuf[HW_DS];
  __shared__ float red[32];
  const int b = blockIdx.x;
  const float* cx = cf + (size_t)b * 2 * HW_DS;
  const float* cy = cx + HW_DS;
  const float* waB = wa + (size_t)b * HW_DS;
  const float* bv = bvec + (size_t)b * HW_DS;
  float* ldB = ld + (size_t)b * HW_DS;

  float x[EP], r[EP], p[EP];
#pragma unroll
  for (int e = 0; e < EP; ++e) {
    const int idx = threadIdx.x + e * 1024;
    x[e] = 0.f;
    if (idx < HW_DS) { x[e] = ldB[idx]; pbuf[idx] = x[e]; }
  }
  __syncthreads();
  float rs_l = 0.f;
#pragma unroll
  for (int e = 0; e < EP; ++e) {
    const int idx = threadIdx.x + e * 1024;
    r[e] = 0.f;
    if (idx < HW_DS) {
      r[e] = bv[idx] - A_apply(pbuf, cx, cy, waB, idx);
      rs_l += r[e] * r[e];
    }
    p[e] = r[e];
  }
  float rs = block_reduce_sum(rs_l, red);
  __syncthreads();
#pragma unroll
  for (int e = 0; e < EP; ++e) {
    const int idx = threadIdx.x + e * 1024;
    if (idx < HW_DS) pbuf[idx] = p[e];
  }
  __syncthreads();

  for (int it = 0; it < iters; ++it) {
    float Ap[EP];
    float pAp_l = 0.f;
#pragma unroll
    for (int e = 0; e < EP; ++e) {
      const int idx = threadIdx.x + e * 1024;
      Ap[e] = 0.f;
      if (idx < HW_DS) {
        Ap[e] = A_apply(pbuf, cx, cy, waB, idx);
        pAp_l += p[e] * Ap[e];
      }
    }
    const float pAp = block_reduce_sum(pAp_l, red);
    const float a = rs / (pAp + 1e-12f);
    float rsn_l = 0.f;
#pragma unroll
    for (int e = 0; e < EP; ++e) {
      x[e] += a * p[e];
      r[e] -= a * Ap[e];
      rsn_l += r[e] * r[e];
    }
    const float rs_n = block_reduce_sum(rsn_l, red);
    const float beta = rs_n / (rs + 1e-12f);
    __syncthreads();
#pragma unroll
    for (int e = 0; e < EP; ++e) {
      const int idx = threadIdx.x + e * 1024;
      p[e] = r[e] + beta * p[e];
      if (idx < HW_DS) pbuf[idx] = p[e];
    }
    __syncthreads();
    rs = rs_n;
  }
#pragma unroll
  for (int e = 0; e < EP; ++e) {
    const int idx = threadIdx.x + e * 1024;
    if (idx < HW_DS) ldB[idx] = x[e];
  }
}

// ---------------- RAFT-style convex upsample + exp ----------------
__global__ void k_upsample(const float* __restrict__ ld, const float* __restrict__ maskb,
                           float* __restrict__ out) {
  int i = blockIdx.x * blockDim.x + threadIdx.x;
  if (i >= B_SZ * H_FULL * W_FULL) return;
  const int b = i / (H_FULL * W_FULL);
  const int rem = i % (H_FULL * W_FULL);
  const int Y = rem / W_FULL, X = rem % W_FULL;
  const int y = Y >> 3, ry = Y & 7, x = X >> 3, rx = X & 7;
  const int pix = y * W_DS + x;
  float lg[9], mx = -1e30f;
#pragma unroll
  for (int k = 0; k < 9; ++k) {
    lg[k] = maskb[((size_t)b * 576 + (k * 64 + ry * 8 + rx)) * HW_DS + pix];
    mx = fmaxf(mx, lg[k]);
  }
  float den = 0.f;
#pragma unroll
  for (int k = 0; k < 9; ++k) { lg[k] = __expf(lg[k] - mx); den += lg[k]; }
  float acc = 0.f;
#pragma unroll
  for (int dy = 0; dy < 3; ++dy)
#pragma unroll
    for (int dx = 0; dx < 3; ++dx) {
      const int yy = y - 1 + dy, xx = x - 1 + dx;
      float dv = 0.f;
      if ((unsigned)yy < (unsigned)H_DS && (unsigned)xx < (unsigned)W_DS)
        dv = ld[(size_t)b * HW_DS + yy * W_DS + xx];
      acc += lg[dy * 3 + dx] * dv;
    }
  out[i] = __expf(acc / den);
}

// ---------------- host orchestration ----------------
extern "C" void kernel_launch(void* const* d_in, const int* in_sizes, int n_in,
                              void* d_out, int out_size, void* d_ws, size_t ws_size,
                              hipStream_t stream) {
  (void)in_sizes; (void)n_in; (void)out_size; (void)ws_size;
  const int B = B_SZ, H = H_FULL, W = W_FULL, h = H_DS, w = W_DS, HW = HW_DS;
  const float* rgb = (const float*)d_in[0];
  const float* dep = (const float*)d_in[1];
  // d_in[2] = K intrinsics (unused by the reference forward)
  const float* Wf[12]; const float* Bf[12];
  for (int i = 0; i < 12; ++i) { Wf[i] = (const float*)d_in[3 + 2 * i]; Bf[i] = (const float*)d_in[4 + 2 * i]; }
  // {Cout, Cin} per conv in params order
  const int WC[12][2] = {{32, 4},  {64, 32},  {128, 64}, {256, 128}, {1, 128},  {64, 3},
                         {128, 320},{128, 320},{128, 320},{2, 128},   {2, 128},  {576, 128}};
  int Kp[12];
  for (int i = 0; i < 12; ++i) Kp[i] = ((WC[i][1] * 9 + 63) / 64) * 64;

  char* base = (char*)d_ws;
  size_t off = 0;
  auto allocf = [&](size_t n) -> float* { float* p = (float*)(base + off); off += ((n * 4 + 255) / 256) * 256; return p; };
  auto allocb = [&](size_t n) -> bf16_t* { bf16_t* p = (bf16_t*)(base + off); off += ((n * 2 + 255) / 256) * 256; return p; };

  bf16_t* Wb[12];
  for (int i = 0; i < 12; ++i) Wb[i] = allocb((size_t)WC[i][0] * Kp[i]);

  float* x0   = allocf((size_t)B * 4 * H * W);
  float* f1   = allocf((size_t)B * 32 * (H / 2) * (W / 2));
  float* f2   = allocf((size_t)B * 64 * (H / 4) * (W / 4));
  float* f3   = allocf((size_t)B * 128 * HW);
  float* ctxb = allocf((size_t)B * 256 * HW);
  float* confb= allocf((size_t)B * HW);
  float* netb = allocf((size_t)B * 128 * HW);
  float* inpb = allocf((size_t)B * 128 * HW);
  float* hxb  = allocf((size_t)B * 320 * HW);
  float* mfeat= allocf((size_t)B * 64 * HW);
  float* zb   = allocf((size_t)B * 128 * HW);
  float* rb   = allocf((size_t)B * 128 * HW);
  float* qb   = allocf((size_t)B * 128 * HW);
  float* dgb  = allocf((size_t)B * 2 * HW);
  float* cfb  = allocf((size_t)B * 2 * HW);
  float* maskb= allocf((size_t)B * 576 * HW);
  float* ldb  = allocf((size_t)B * HW);
  float* gb   = allocf((size_t)B * 2 * HW);
  float* mdsb = allocf((size_t)B * HW);
  float* slogb= allocf((size_t)B * HW);
  float* wab  = allocf((size_t)B * HW);
  float* bvecb= allocf((size_t)B * HW);
  float* meanb= allocf((size_t)B);
  float* menci= allocf((size_t)B * 3 * HW);

  auto cdiv = [](int a, int b2) { return (a + b2 - 1) / b2; };

  // bf16 weight conversion + fragment permutation + K zero-padding (cheap, once per call)
  for (int i = 0; i < 12; ++i) {
    int n = WC[i][0] * Kp[i];
    k_wperm<<<cdiv(n, 256), 256, 0, stream>>>(Wf[i], Wb[i], WC[i][0], WC[i][1] * 9, Kp[i]);
  }

  auto conv = [&](const float* in, int widx, float* out, int Cin, int Cout,
                  int Hin, int Win, int stride, int act) {
    const int Hout = (Hin + stride - 1) / stride, Wout = (Win + stride - 1) / stride;
    const int M = Hout * Wout;
    dim3 g(cdiv(M, 128), cdiv(Cout, 64), B), blk(256);
    const int padlo = (stride == 2) ? 0 : 1;  // lax 'SAME': pad_lo = pad_total//2
    switch (act) {
      case 0: k_conv_wmma<0><<<g, blk, 0, stream>>>(in, Wb[widx], Bf[widx], out, Cin, Cout, Hin, Win, Hout, Wout, stride, padlo, Kp[widx]); break;
      case 1: k_conv_wmma<1><<<g, blk, 0, stream>>>(in, Wb[widx], Bf[widx], out, Cin, Cout, Hin, Win, Hout, Wout, stride, padlo, Kp[widx]); break;
      case 2: k_conv_wmma<2><<<g, blk, 0, stream>>>(in, Wb[widx], Bf[widx], out, Cin, Cout, Hin, Win, Hout, Wout, stride, padlo, Kp[widx]); break;
      default:k_conv_wmma<3><<<g, blk, 0, stream>>>(in, Wb[widx], Bf[widx], out, Cin, Cout, Hin, Win, Hout, Wout, stride, padlo, Kp[widx]); break;
    }
  };

  // ---- encoder ----
  k_concat4<<<cdiv(B * 4 * H * W, 256), 256, 0, stream>>>(rgb, dep, x0, H * W);
  k_downsample<<<cdiv(B * HW, 256), 256, 0, stream>>>(dep, mdsb, slogb);
  conv(x0, 0, f1, 4, 32, H, W, 2, 1);
  conv(f1, 1, f2, 32, 64, H / 2, W / 2, 2, 1);
  conv(f2, 2, f3, 64, 128, H / 4, W / 4, 2, 1);
  conv(f3, 3, ctxb, 128, 256, h, w, 1, 0);
  conv(f3, 4, confb, 128, 1, h, w, 1, 2);
  k_splitctx<<<cdiv(B * 128 * HW, 256), 256, 0, stream>>>(ctxb, netb, inpb);
  k_wa<<<cdiv(B * HW, 256), 256, 0, stream>>>(mdsb, confb, wab, B * HW);
  k_zero<<<cdiv(B * HW, 256), 256, 0, stream>>>(ldb, B * HW);
  k_zero<<<cdiv(B * 2 * HW, 256), 256, 0, stream>>>(gb, B * 2 * HW);

  // ---- GRU iterations ----
  for (int it = 0; it < GRU_ITERS; ++it) {
    k_mean<<<B, 256, 0, stream>>>(ldb, meanb);
    k_mencin<<<cdiv(B * HW, 256), 256, 0, stream>>>(ldb, gb, meanb, menci);
    conv(menci, 5, mfeat, 3, 64, h, w, 1, 1);
    k_hx<<<cdiv(B * 320 * HW, 256), 256, 0, stream>>>(netb, inpb, mfeat, hxb);
    conv(hxb, 6, zb, 320, 128, h, w, 1, 2);
    conv(hxb, 7, rb, 320, 128, h, w, 1, 2);
    k_rnet<<<cdiv(B * 128 * HW, 256), 256, 0, stream>>>(hxb, rb, netb);
    conv(hxb, 8, qb, 320, 128, h, w, 1, 3);
    k_gru<<<cdiv(B * 128 * HW, 256), 256, 0, stream>>>(netb, zb, qb, B * 128 * HW);
    conv(netb, 9, dgb, 128, 2, h, w, 1, 0);
    conv(netb, 10, cfb, 128, 2, h, w, 1, 0);
    conv(netb, 11, maskb, 128, 576, h, w, 1, 0);
    k_post<<<cdiv(B * 2 * HW, 256), 256, 0, stream>>>(gb, dgb, cfb, B * 2 * HW);
    k_buildb<<<cdiv(B * HW, 256), 256, 0, stream>>>(cfb, gb, wab, slogb, bvecb);
    k_cg<<<B, 1024, 0, stream>>>(cfb, wab, bvecb, ldb, CG_ITERS_C);
  }

  // ---- final convex upsample + exp (only last iteration's pred is returned) ----
  k_upsample<<<cdiv(B * H * W, 256), 256, 0, stream>>>(ldb, maskb, (float*)d_out);
}
